// MultiHeadLatentAttention_3796751090026
// MI455X (gfx1250) — matrix operand order repaired and checked by reading, not compiled, not run on hardware
//
#include <hip/hip_runtime.h>
#include <stdint.h>

// ---------------------------------------------------------------------------
// MLA forward for MI455X (gfx1250, wave32).
// GEMMs + attention on v_wmma_f32_16x16x32_bf16; A-tiles staged to LDS with
// async-to-LDS loads (ASYNCcnt double-buffer); prefetch for attention tiles.
// ---------------------------------------------------------------------------

typedef __attribute__((ext_vector_type(16))) __bf16 v16bf;
typedef __attribute__((ext_vector_type(8)))  float  v8f;
typedef __attribute__((ext_vector_type(4)))  int    v4i;
typedef unsigned short bf16_t;

#define AS1 __attribute__((address_space(1)))
#define AS3 __attribute__((address_space(3)))

#define HIDDEN   2048
#define NUMHEADS 16
#define HEADDIM  128
#define KVC      512
#define QC       1536
#define ROPEDIM  64
#define BB       2
#define SS       2048
#define QKD      192   // HEADDIM + ROPEDIM
#define MTOK     (BB*SS)

__device__ __forceinline__ bf16_t f2b(float f) {
    union { float f; unsigned u; } x; x.f = f;
    unsigned r = x.u + 0x7FFFu + ((x.u >> 16) & 1u);   // round-to-nearest-even
    return (bf16_t)(r >> 16);
}
__device__ __forceinline__ float b2f(bf16_t b) {
    union { unsigned u; float f; } x; x.u = ((unsigned)b) << 16;
    return x.f;
}

union Frag { v16bf v; uint4 q[2]; };

// ---------------------------------------------------------------------------
// CDNA5 async global->LDS (ASYNCcnt-tracked). Builtin exists on this toolchain:
// (AS1 v4i* gsrc, AS3 v4i* ldsdst, int offset, int cpol).
// ---------------------------------------------------------------------------
__device__ __forceinline__ void async_load_b128(const bf16_t* g, bf16_t* l) {
#if defined(__has_builtin) && __has_builtin(__builtin_amdgcn_global_load_async_to_lds_b128)
    __builtin_amdgcn_global_load_async_to_lds_b128((AS1 v4i*)g, (AS3 v4i*)l, 0, 0);
#else
    unsigned loff = (unsigned)(uintptr_t)(AS3 void*)l;
    asm volatile("global_load_async_to_lds_b128 %0, %1, off"
                 :: "v"(loff), "v"((unsigned long long)(uintptr_t)g)
                 : "memory");
#endif
}
__device__ __forceinline__ void wait_async_le2() {
#if defined(__has_builtin) && __has_builtin(__builtin_amdgcn_s_wait_asynccnt)
    __builtin_amdgcn_s_wait_asynccnt(2);
#else
    asm volatile("s_wait_asynccnt 2" ::: "memory");
#endif
}
__device__ __forceinline__ void wait_async_le0() {
#if defined(__has_builtin) && __has_builtin(__builtin_amdgcn_s_wait_asynccnt)
    __builtin_amdgcn_s_wait_asynccnt(0);
#else
    asm volatile("s_wait_asynccnt 0" ::: "memory");
#endif
}

// A-fragment 16x32 (M x K), row-major source, stride in bf16 elements.
// Lane L: row M = L%16; chunk0 halves K = 8*(L/16)+0..7, chunk1 K = 16+8*(L/16)+0..7.
__device__ __forceinline__ v16bf load_frag_a(const bf16_t* __restrict__ base, int stride) {
    Frag f;
    const int lane = threadIdx.x & 31;
    const bf16_t* p = base + (size_t)(lane & 15) * stride + ((lane >> 4) << 3);
    f.q[0] = *(const uint4*)(p);
    f.q[1] = *(const uint4*)(p + 16);
    return f.v;
}

// B-fragment 32x16 (K x N), source is B^T row-major ([N, K]), stride in bf16 elems.
// Lane L: col N = L%16; chunk0 K = 8*(L/16)+0..7, chunk1 K = 16+8*(L/16)+0..7.
__device__ __forceinline__ v16bf load_frag_b(const bf16_t* __restrict__ base, int stride) {
    Frag f;
    const int lane = threadIdx.x & 31;
    const bf16_t* p = base + (size_t)(lane & 15) * stride + ((lane >> 4) << 3);
    f.q[0] = *(const uint4*)(p);
    f.q[1] = *(const uint4*)(p + 16);
    return f.v;
}

__device__ __forceinline__ v8f wmma_bf16(v16bf a, v16bf b, v8f c) {
    return __builtin_amdgcn_wmma_f32_16x16x32_bf16(false, a, false, b, (short)0, c,
                                                   false, false);
}

// ---------------------------------------------------------------------------
// elementwise fp32 -> bf16
// ---------------------------------------------------------------------------
__global__ void k_f32_to_bf16(const float* __restrict__ in, bf16_t* __restrict__ out,
                              size_t n) {
    size_t i = (size_t)blockIdx.x * blockDim.x + threadIdx.x;
    if (i < n) out[i] = f2b(in[i]);
}

// ---------------------------------------------------------------------------
// transpose + convert weight: W [K,N] fp32  ->  WT [N,K] bf16 (LDS tiled)
// K, N multiples of 32. block (32,8), grid (N/32, K/32)
// ---------------------------------------------------------------------------
__global__ void k_transpose_bf16(const float* __restrict__ W, bf16_t* __restrict__ WT,
                                 int K, int N) {
    __shared__ float tile[32][33];
    const int kb = blockIdx.y * 32, nb = blockIdx.x * 32;
    const int tx = threadIdx.x, ty = threadIdx.y;
    for (int i = 0; i < 32; i += 8)
        tile[ty + i][tx] = W[(size_t)(kb + ty + i) * N + nb + tx];
    __syncthreads();
    for (int i = 0; i < 32; i += 8)
        WT[(size_t)(nb + ty + i) * K + kb + tx] = f2b(tile[tx][ty + i]);
}

// ---------------------------------------------------------------------------
// GEMM: C[M,N] = A[M,K](bf16,row-major) x W (given as WT[N,K] bf16) + bias(f32)
// block 256 = 8 waves; block tile 128x128; wave tile 64(M) x 32(N).
// A block tile (128x32) double-buffered in LDS via async-to-LDS loads.
// grid (N/128, M/128).
// ---------------------------------------------------------------------------
template <bool OUT_BF16>
__global__ __launch_bounds__(256) void k_gemm_bf16(
    const bf16_t* __restrict__ A, const bf16_t* __restrict__ WT,
    const float* __restrict__ bias, void* __restrict__ C,
    int M, int N, int K) {
    __shared__ alignas(16) bf16_t Atile[2][128][32];

    const int tid  = threadIdx.x;
    const int wave = tid >> 5;
    const int lane = tid & 31;
    const int wm = wave >> 2;          // 0..1
    const int wn = wave & 3;           // 0..3
    const int rowblk = blockIdx.y * 128;
    const int row0 = rowblk + wm * 64;
    const int col0 = blockIdx.x * 128 + wn * 32;

    // async fill of one 128x32 A tile: 256 threads x 2 b128 each
    const int f_row  = tid >> 1;             // 0..127
    const int f_coff = (tid & 1) * 16;       // half-offset 0 or 16
    auto fill = [&](int bb, int k0) {
        const bf16_t* g = A + (size_t)(rowblk + f_row) * K + k0 + f_coff;
        async_load_b128(g,     &Atile[bb][f_row][f_coff]);
        async_load_b128(g + 8, &Atile[bb][f_row][f_coff + 8]);
    };

    v8f acc[4][2];
    const float b0 = bias[col0 + (lane & 15)];
    const float b1 = bias[col0 + 16 + (lane & 15)];
    for (int i = 0; i < 4; ++i)
        for (int r = 0; r < 8; ++r) { acc[i][0][r] = b0; acc[i][1][r] = b1; }

    const int nk = K >> 5;
    fill(0, 0);
    for (int i = 0; i < nk; ++i) {
        const int cur = i & 1;
        if (i + 1 < nk) { fill(1 - cur, (i + 1) << 5); wait_async_le2(); }
        else            { wait_async_le0(); }
        __syncthreads();                 // current tile resident for all waves

        v16bf a[4], b[2];
        for (int t = 0; t < 4; ++t) {    // A-frags from LDS (row stride 32 halves)
            const bf16_t* p = &Atile[cur][wm * 64 + 16 * t + (lane & 15)][(lane >> 4) << 3];
            Frag f;
            f.q[0] = *(const uint4*)(p);
            f.q[1] = *(const uint4*)(p + 16);
            a[t] = f.v;
        }
        b[0] = load_frag_b(WT + (size_t)col0 * K + (i << 5), K);
        b[1] = load_frag_b(WT + (size_t)(col0 + 16) * K + (i << 5), K);
        for (int t = 0; t < 4; ++t) {
            acc[t][0] = wmma_bf16(a[t], b[0], acc[t][0]);
            acc[t][1] = wmma_bf16(a[t], b[1], acc[t][1]);
        }
        __syncthreads();                 // everyone done reading before overwrite
    }

    const int cm = lane >> 4, cn = lane & 15;
    for (int i = 0; i < 4; ++i)
        for (int j = 0; j < 2; ++j)
            for (int r = 0; r < 8; ++r) {
                const int m = row0 + 16 * i + r + 8 * cm;
                const int n = col0 + 16 * j + cn;
                const float v = acc[i][j][r];
                if (OUT_BF16) ((bf16_t*)C)[(size_t)m * N + n] = f2b(v);
                else          ((float*)C)[(size_t)m * N + n]  = v;
            }
}

// ---------------------------------------------------------------------------
// pack Q/K: c-part [B*S, H*128] + r-part [B*S, H*64] (rope applied)
//   -> out [B, H, S, 192].  block 192, grid (S, H, B)
// ---------------------------------------------------------------------------
__global__ void k_pack_qk_rope(const bf16_t* __restrict__ cpart,
                               const bf16_t* __restrict__ rpart,
                               bf16_t* __restrict__ out) {
    const int t = threadIdx.x;
    const int s = blockIdx.x, h = blockIdx.y, b = blockIdx.z;
    const size_t tok = (size_t)b * SS + s;
    const size_t o = (((size_t)b * NUMHEADS + h) * SS + s) * QKD;
    if (t < HEADDIM) {
        out[o + t] = cpart[tok * (NUMHEADS * HEADDIM) + h * HEADDIM + t];
    } else {
        const int d = t - HEADDIM;        // 0..63
        const int i = d >> 1;
        // inv_freq = 10000^(-2i/64); ln(10000)=9.210340371976184
        const float inv = __expf(-((float)(2 * i) * (1.0f / 64.0f)) * 9.2103403719761836f);
        const float ang = (float)s * inv;
        const float sn = __sinf(ang), cs = __cosf(ang);
        const float x1 = b2f(rpart[tok * (NUMHEADS * ROPEDIM) + h * ROPEDIM + 2 * i]);
        const float x2 = b2f(rpart[tok * (NUMHEADS * ROPEDIM) + h * ROPEDIM + 2 * i + 1]);
        const float v = (d & 1) ? (x1 * sn + x2 * cs) : (x1 * cs - x2 * sn);
        out[o + HEADDIM + d] = f2b(v);
    }
}

// ---------------------------------------------------------------------------
// pack V transposed: vals [B*S, H*128] -> Vt [B, H, 128, S] (LDS-tiled transpose)
// block (32,8), grid (S/32, 128/32, B*H)
// ---------------------------------------------------------------------------
__global__ void k_pack_vt(const bf16_t* __restrict__ vals, bf16_t* __restrict__ vt) {
    __shared__ bf16_t tile[32][33];
    const int bh = blockIdx.z;
    const int b = bh >> 4, h = bh & 15;
    const int s0 = blockIdx.x * 32, d0 = blockIdx.y * 32;
    const int tx = threadIdx.x, ty = threadIdx.y;
    for (int i = 0; i < 32; i += 8)
        tile[ty + i][tx] =
            vals[((size_t)b * SS + s0 + ty + i) * (NUMHEADS * HEADDIM) + h * HEADDIM + d0 + tx];
    __syncthreads();
    for (int i = 0; i < 32; i += 8)
        vt[((size_t)bh * HEADDIM + d0 + ty + i) * SS + s0 + tx] = tile[tx][ty + i];
}

// ---------------------------------------------------------------------------
// Flash attention (causal). One wave = 16 query rows. block 256 = 8 waves.
// grid (S/128, H, B).   Q,K: [B,H,S,192] bf16; Vt: [B,H,128,S] bf16.
// ctx out: [B*S, H*128] bf16 (token-major for the output GEMM).
// ---------------------------------------------------------------------------
__global__ __launch_bounds__(256) void k_mla_attention(
    const bf16_t* __restrict__ Q, const bf16_t* __restrict__ K,
    const bf16_t* __restrict__ Vt, bf16_t* __restrict__ ctx) {
    __shared__ alignas(16) bf16_t plds[8][16][32];   // per-wave P staging (16x32)

    const int wave = threadIdx.x >> 5;
    const int lane = threadIdx.x & 31;
    const int h = blockIdx.y, b = blockIdx.z;
    const int q0 = (blockIdx.x * 8 + wave) * 16;

    const bf16_t* Qbh = Q + ((size_t)b * NUMHEADS + h) * SS * QKD;
    const bf16_t* Kbh = K + ((size_t)b * NUMHEADS + h) * SS * QKD;
    const bf16_t* Vbh = Vt + ((size_t)b * NUMHEADS + h) * HEADDIM * (size_t)SS;

    // Q fragments resident: 16 rows x 192 = 6 A-frags
    v16bf aq[6];
    for (int kk = 0; kk < 6; ++kk)
        aq[kk] = load_frag_a(Qbh + (size_t)q0 * QKD + 32 * kk, QKD);

    v8f acc[8];                                  // context 16 x 128
    for (int t = 0; t < 8; ++t)
        for (int r = 0; r < 8; ++r) acc[t][r] = 0.0f;
    float mrun[8], lrun[8];
    for (int r = 0; r < 8; ++r) { mrun[r] = -1e30f; lrun[r] = 0.0f; }

    const float scale = 0.072168783648703332f;   // 1/sqrt(192)
    const int cm = lane >> 4, cn = lane & 15;

    for (int j0 = 0; j0 < q0 + 16; j0 += 32) {   // causal key-tile loop (32 keys/iter)
        // warm L0 for the next key/value tiles while this tile computes
        if (j0 + 32 < q0 + 16) {
            __builtin_prefetch(Kbh + (size_t)(j0 + 32) * QKD + lane * 192, 0, 3);
            __builtin_prefetch(Vbh + (size_t)(lane * 4) * SS + (j0 + 32), 0, 3);
        }
        v8f s0, s1;
        for (int r = 0; r < 8; ++r) { s0[r] = 0.0f; s1[r] = 0.0f; }
        for (int kk = 0; kk < 6; ++kk) {
            v16bf bk0 = load_frag_b(Kbh + (size_t)j0 * QKD + 32 * kk, QKD);
            v16bf bk1 = load_frag_b(Kbh + (size_t)(j0 + 16) * QKD + 32 * kk, QKD);
            s0 = wmma_bf16(aq[kk], bk0, s0);
            s1 = wmma_bf16(aq[kk], bk1, s1);
        }
        // scale + causal mask + online softmax (row groups = 16-lane halves)
        for (int r = 0; r < 8; ++r) {
            const int qrow = q0 + r + 8 * cm;
            float v0 = s0[r] * scale;
            float v1 = s1[r] * scale;
            if (j0 + cn > qrow)      v0 = -1e30f;
            if (j0 + 16 + cn > qrow) v1 = -1e30f;
            float rmax = fmaxf(v0, v1);
            for (int off = 1; off < 16; off <<= 1)
                rmax = fmaxf(rmax, __shfl_xor(rmax, off, 32));
            const float newm = fmaxf(mrun[r], rmax);
            const float alpha = __expf(mrun[r] - newm);
            mrun[r] = newm;
            const float p0 = __expf(v0 - newm);
            const float p1 = __expf(v1 - newm);
            float rs = p0 + p1;
            for (int off = 1; off < 16; off <<= 1)
                rs += __shfl_xor(rs, off, 32);
            lrun[r] = lrun[r] * alpha + rs;
            for (int t = 0; t < 8; ++t) acc[t][r] *= alpha;
            // stage P row-major into per-wave LDS tile (C-layout -> memory)
            const int m = r + 8 * cm;
            plds[wave][m][cn]      = f2b(p0);
            plds[wave][m][16 + cn] = f2b(p1);
        }
        // reload P as an A-fragment (lane transpose via LDS; compiler inserts dscnt waits)
        Frag pf;
        {
            const bf16_t* p = &plds[wave][lane & 15][(lane >> 4) << 3];
            pf.q[0] = *(const uint4*)(p);
            pf.q[1] = *(const uint4*)(p + 16);
        }
        // context += P(16x32) x V(32x128)
        for (int t = 0; t < 8; ++t) {
            v16bf bv = load_frag_b(Vbh + (size_t)(16 * t) * SS + j0, SS);
            acc[t] = wmma_bf16(pf.v, bv, acc[t]);
        }
    }

    // finalize: divide by l, write ctx token-major [B*S, H*128]
    for (int r = 0; r < 8; ++r) {
        const float inv = 1.0f / lrun[r];
        const int qrow = q0 + r + 8 * cm;
        const size_t base = ((size_t)b * SS + qrow) * (NUMHEADS * HEADDIM) + h * HEADDIM;
        for (int t = 0; t < 8; ++t)
            ctx[base + 16 * t + cn] = f2b(acc[t][r] * inv);
    }
}

// ---------------------------------------------------------------------------
// host-side launch
// ---------------------------------------------------------------------------
extern "C" void kernel_launch(void* const* d_in, const int* in_sizes, int n_in,
                              void* d_out, int out_size, void* d_ws, size_t ws_size,
                              hipStream_t stream) {
    (void)in_sizes; (void)n_in; (void)out_size; (void)ws_size;

    const float* x      = (const float*)d_in[0];
    // d_in[1] = attention_mask (all ones -> causal only; ignored)
    const float* w_kvd  = (const float*)d_in[2];
    const float* b_kvd  = (const float*)d_in[3];
    const float* w_ku   = (const float*)d_in[4];
    const float* b_ku   = (const float*)d_in[5];
    const float* w_vu   = (const float*)d_in[6];
    const float* b_vu   = (const float*)d_in[7];
    const float* w_kr   = (const float*)d_in[8];
    const float* b_kr   = (const float*)d_in[9];
    const float* w_qd   = (const float*)d_in[10];
    const float* b_qd   = (const float*)d_in[11];
    const float* w_qu   = (const float*)d_in[12];
    const float* b_qu   = (const float*)d_in[13];
    const float* w_qr   = (const float*)d_in[14];
    const float* b_qr   = (const float*)d_in[15];
    const float* w_o    = (const float*)d_in[16];
    const float* b_o    = (const float*)d_in[17];
    float* out          = (float*)d_out;

    char* ws = (char*)d_ws;
    size_t off = 0;
    auto take = [&](size_t bytes) -> size_t {
        size_t o = off; off += (bytes + 255) & ~(size_t)255; return o;
    };
    const size_t o_xb   = take((size_t)MTOK * HIDDEN * 2);        // x bf16; reused for q_up
    const size_t o_wkvd = take((size_t)KVC * HIDDEN * 2);
    const size_t o_wku  = take((size_t)(NUMHEADS*HEADDIM) * KVC * 2);
    const size_t o_wvu  = take((size_t)(NUMHEADS*HEADDIM) * KVC * 2);
    const size_t o_wkr  = take((size_t)(NUMHEADS*ROPEDIM) * KVC * 2);
    const size_t o_wqd  = take((size_t)QC * HIDDEN * 2);
    const size_t o_wqu  = take((size_t)(NUMHEADS*HEADDIM) * QC * 2);
    const size_t o_wqr  = take((size_t)(NUMHEADS*ROPEDIM) * QC * 2);
    const size_t o_wo   = take((size_t)HIDDEN * (NUMHEADS*HEADDIM) * 2);
    const size_t o_kvc  = take((size_t)MTOK * KVC * 2);
    const size_t o_qc   = take((size_t)MTOK * QC * 2);
    const size_t o_kct  = take((size_t)MTOK * (NUMHEADS*HEADDIM) * 2); // keys_c; reused as ctx
    const size_t o_vtt  = take((size_t)MTOK * (NUMHEADS*HEADDIM) * 2);
    const size_t o_krt  = take((size_t)MTOK * (NUMHEADS*ROPEDIM) * 2);
    const size_t o_qrt  = take((size_t)MTOK * (NUMHEADS*ROPEDIM) * 2);
    const size_t o_Q    = take((size_t)BB * NUMHEADS * SS * QKD * 2);
    const size_t o_K    = take((size_t)BB * NUMHEADS * SS * QKD * 2);
    const size_t o_V    = take((size_t)BB * NUMHEADS * HEADDIM * SS * 2);

    bf16_t* xb   = (bf16_t*)(ws + o_xb);
    bf16_t* qupt = (bf16_t*)(ws + o_xb);     // alias: xb dead after down-projections
    bf16_t* kvc  = (bf16_t*)(ws + o_kvc);
    bf16_t* qc   = (bf16_t*)(ws + o_qc);
    bf16_t* kct  = (bf16_t*)(ws + o_kct);
    bf16_t* ctx  = (bf16_t*)(ws + o_kct);    // alias: keys_c dead after K pack
    bf16_t* vtt  = (bf16_t*)(ws + o_vtt);
    bf16_t* krt  = (bf16_t*)(ws + o_krt);
    bf16_t* qrt  = (bf16_t*)(ws + o_qrt);
    bf16_t* Qb   = (bf16_t*)(ws + o_Q);
    bf16_t* Kb   = (bf16_t*)(ws + o_K);
    bf16_t* Vb   = (bf16_t*)(ws + o_V);

    // 1) convert x -> bf16
    {
        size_t n = (size_t)MTOK * HIDDEN;
        k_f32_to_bf16<<<dim3((unsigned)((n + 255) / 256)), dim3(256), 0, stream>>>(x, xb, n);
    }
    // 2) transpose-convert all weights to [N,K] bf16
    auto tr = [&](const float* W, size_t o, int K, int N) {
        k_transpose_bf16<<<dim3(N / 32, K / 32), dim3(32, 8), 0, stream>>>(
            W, (bf16_t*)(ws + o), K, N);
    };
    tr(w_kvd, o_wkvd, HIDDEN, KVC);
    tr(w_ku,  o_wku,  KVC, NUMHEADS * HEADDIM);
    tr(w_vu,  o_wvu,  KVC, NUMHEADS * HEADDIM);
    tr(w_kr,  o_wkr,  KVC, NUMHEADS * ROPEDIM);
    tr(w_qd,  o_wqd,  HIDDEN, QC);
    tr(w_qu,  o_wqu,  QC, NUMHEADS * HEADDIM);
    tr(w_qr,  o_wqr,  QC, NUMHEADS * ROPEDIM);
    tr(w_o,   o_wo,   NUMHEADS * HEADDIM, HIDDEN);

    auto gemm_b = [&](const bf16_t* A, size_t oWT, const float* bias, bf16_t* C,
                      int N, int K) {
        k_gemm_bf16<true><<<dim3(N / 128, MTOK / 128), dim3(256), 0, stream>>>(
            A, (const bf16_t*)(ws + oWT), bias, (void*)C, MTOK, N, K);
    };

    // 3) down projections from x
    gemm_b(xb, o_wkvd, b_kvd, kvc, KVC, HIDDEN);
    gemm_b(xb, o_wqd,  b_qd,  qc,  QC,  HIDDEN);
    // 4) up projections (xb now dead -> q_up aliases it)
    gemm_b(kvc, o_wku, b_ku, kct, NUMHEADS * HEADDIM, KVC);
    gemm_b(kvc, o_wvu, b_vu, vtt, NUMHEADS * HEADDIM, KVC);
    gemm_b(kvc, o_wkr, b_kr, krt, NUMHEADS * ROPEDIM, KVC);
    gemm_b(qc,  o_wqu, b_qu, qupt, NUMHEADS * HEADDIM, QC);
    gemm_b(qc,  o_wqr, b_qr, qrt,  NUMHEADS * ROPEDIM, QC);

    // 5) pack + rope
    k_pack_qk_rope<<<dim3(SS, NUMHEADS, BB), dim3(QKD), 0, stream>>>(qupt, qrt, Qb);
    k_pack_qk_rope<<<dim3(SS, NUMHEADS, BB), dim3(QKD), 0, stream>>>(kct, krt, Kb);
    k_pack_vt<<<dim3(SS / 32, HEADDIM / 32, BB * NUMHEADS), dim3(32, 8), 0, stream>>>(vtt, Vb);

    // 6) causal flash attention (WMMA) -> ctx (aliases keys_c buffer, now dead)
    k_mla_attention<<<dim3(SS / 128, NUMHEADS, BB), dim3(256), 0, stream>>>(Qb, Kb, Vb, ctx);

    // 7) output projection -> fp32 d_out
    k_gemm_bf16<false><<<dim3(HIDDEN / 128, MTOK / 128), dim3(256), 0, stream>>>(
        ctx, (const bf16_t*)(ws + o_wo), b_o, (void*)out, MTOK, HIDDEN,
        NUMHEADS * HEADDIM);
}
